// MaceNet_89885075571145
// MI455X (gfx1250) — compile-verified, matrix operand order checked by reading, not run on hardware
//
#include <hip/hip_runtime.h>
#include <hip/hip_bf16.h>
#include <math.h>

// ---------------------------------------------------------------------------
// MACE 2-layer message passing on MI455X (gfx1250, wave32, WMMA).
//   N=50000 nodes, E=800000 edges, A=4, R=8, C=64, SH=4, H=256, G=16
// Dense GEMMs (50000x256x{64,256}) on v_wmma_f32_16x16x32_f16 (f32 acc).
// Edge scatter uses native f32 global atomics into an L2-resident (51MB) acc.
// ---------------------------------------------------------------------------

typedef __attribute__((ext_vector_type(16))) _Float16 v16h;
typedef __attribute__((ext_vector_type(8)))  float    v8f;
typedef __attribute__((ext_vector_type(4)))  float    v4f;

#define ADIM  4
#define RDIM  8
#define CDIM  64
#define SHDIM 4
#define HDIM  256
#define GDIM  16

// ---------------------------------------------------------------------------
// Edge spherical harmonics (lmax=1): sh = [1, vx/r, vy/r, vz/r]
// ---------------------------------------------------------------------------
__global__ __launch_bounds__(256) void k_edge_sh(
    const float* __restrict__ pos, const int* __restrict__ ei,
    float* __restrict__ sh, int E)
{
    int e = blockIdx.x * 256 + threadIdx.x;
    if (e >= E) return;
    int s = ei[e];
    int d = ei[E + e];
    float vx = pos[(size_t)d * 3 + 0] - pos[(size_t)s * 3 + 0];
    float vy = pos[(size_t)d * 3 + 1] - pos[(size_t)s * 3 + 1];
    float vz = pos[(size_t)d * 3 + 2] - pos[(size_t)s * 3 + 2];
    float r  = sqrtf(vx * vx + vy * vy + vz * vz);
    float inv = 1.0f / (r + 1e-8f);
    v4f o; o.x = 1.0f; o.y = vx * inv; o.z = vy * inv; o.w = vz * inv;
    *(v4f*)(sh + (size_t)e * 4) = o;
}

// ---------------------------------------------------------------------------
// feats = x @ W_feat1 + b  (K=4 -> plain VALU, WMMA would be wasted)
// ---------------------------------------------------------------------------
__global__ __launch_bounds__(256) void k_feats1(
    const float* __restrict__ x, const float* __restrict__ Wf,
    const float* __restrict__ b, float* __restrict__ feats, int total)
{
    int idx = blockIdx.x * 256 + threadIdx.x;
    if (idx >= total) return;
    int n = idx >> 6, c = idx & 63;
    float s = b[c];
#pragma unroll
    for (int a = 0; a < ADIM; ++a)
        s = fmaf(x[(size_t)n * ADIM + a], Wf[a * CDIM + c], s);
    feats[idx] = s;
}

// ---------------------------------------------------------------------------
// h[n,j] += sum_a x[n,a] * W_skip1[a,j]   (A=4 skip connection)
// ---------------------------------------------------------------------------
__global__ __launch_bounds__(256) void k_skip4(
    const float* __restrict__ x, const float* __restrict__ Ws,
    float* __restrict__ h, int total)
{
    int idx = blockIdx.x * 256 + threadIdx.x;
    if (idx >= total) return;
    int n = idx >> 8, j = idx & 255;
    float s = h[idx];
#pragma unroll
    for (int a = 0; a < ADIM; ++a)
        s = fmaf(x[(size_t)n * ADIM + a], Ws[a * HDIM + j], s);
    h[idx] = s;
}

// ---------------------------------------------------------------------------
// Edge message + scatter: one wave per edge, 2 channels per lane.
//   rad_c = silu(edge_attr[e] . W_rad[:,c]); m = feats[src,c] * rad_c
//   atomic agg[dst, c*4 + k] += m * sh[e,k]
// agg is 51MB -> L2 resident; unsafeAtomicAdd -> global_atomic_add_f32.
// ---------------------------------------------------------------------------
__global__ __launch_bounds__(256) void k_scatter(
    const float* __restrict__ feats, const float* __restrict__ ea,
    const float* __restrict__ Wrad,  const float* __restrict__ sh,
    const int* __restrict__ srcI,    const int* __restrict__ dstI,
    float* __restrict__ agg, int E)
{
    __shared__ float sWr[RDIM * CDIM];           // 2KB radial weights
    int tid = threadIdx.x;
    for (int i = tid; i < RDIM * CDIM; i += 256) sWr[i] = Wrad[i];
    __syncthreads();

    int wave = tid >> 5, lane = tid & 31;
    int e = blockIdx.x * 8 + wave;
    if (e >= E) return;

    int s = srcI[e], d = dstI[e];
    __builtin_prefetch(feats + (size_t)s * CDIM, 0, 0);   // global_prefetch
    v4f sh4 = *(const v4f*)(sh + (size_t)e * 4);
    v4f e0  = *(const v4f*)(ea + (size_t)e * RDIM);
    v4f e1  = *(const v4f*)(ea + (size_t)e * RDIM + 4);

#pragma unroll
    for (int u = 0; u < 2; ++u) {
        int c = lane * 2 + u;
        float acc = 0.0f;
#pragma unroll
        for (int r = 0; r < 4; ++r) acc = fmaf(e0[r], sWr[r * CDIM + c], acc);
#pragma unroll
        for (int r = 0; r < 4; ++r) acc = fmaf(e1[r], sWr[(r + 4) * CDIM + c], acc);
        float radv = acc / (1.0f + __expf(-acc));        // silu
        float m = feats[(size_t)s * CDIM + c] * radv;
        float* ap = agg + (size_t)d * HDIM + c * SHDIM;
#pragma unroll
        for (int k = 0; k < SHDIM; ++k)
            unsafeAtomicAdd(ap + k, m * sh4[k]);
    }
}

// ---------------------------------------------------------------------------
// WMMA GEMM: Y(N x J) = f16(X(N x 256)) * f16(W(256 x J)) [+ bias][+ Csrc]
//
// Block = 256 threads = 8 waves, handles ROWT row tiles (ROWT*16 rows).
// J=256, NCT=2: wave w owns column tiles {w, w+8}
// J=64,  NCT=1: waves 0..3 own one column tile; 4..7 help stage LDS only.
// Per K-chunk: W(32 x J) chunk staged to LDS as f16 [j][k] (pair-packed dword
// stores, b128 global reads); B-fragments read once and reused across all
// ROWT A-fragments -> ROWT*NCT WMMAs per wave per chunk.
//
// All fragment/accumulator indices are compile-time constants after full
// unrolling (no v_movrel dynamic VGPR indexing).
//
// A-fragment per 16-bit ISA layout: lane l, m = l&15, hi = l>>4:
//   halves 0..7  = X[m, kc+hi*8   .. +7]   (two b128 loads)
//   halves 8..15 = X[m, kc+16+hi*8.. +7]   (two b128 loads)
// C/D layout: lane n = l&15, vgpr r -> m = r + 8*hi.
// ---------------------------------------------------------------------------
template <int J, int NCT, int ROWT>
__global__ __launch_bounds__(256) void k_gemm_wmma(
    const float* __restrict__ X, const float* __restrict__ W,
    const float* __restrict__ bias, const float* __restrict__ Csrc,
    float* __restrict__ Y, int rowTiles)
{
    constexpr int K = 256;
    constexpr int colTiles = J >> 4;
    __shared__ unsigned sWu[J * 16];             // f16 chunk, [j][k], dword-packed
    const _Float16* sW = (const _Float16*)sWu;

    const int tid  = threadIdx.x;
    const int wave = tid >> 5, lane = tid & 31;
    const int hi   = lane >> 4, ln = lane & 15;
    const int rtb  = blockIdx.x * ROWT;

    const bool active = (NCT == 2) ? true : (wave < colTiles);
    const int col0 = wave * 16 + ln;             // ct0 = wave
    const int col1 = (wave + 8) * 16 + ln;       // ct1 = wave + 8 (NCT==2)

    bool valid[ROWT];
#pragma unroll
    for (int i = 0; i < ROWT; ++i) valid[i] = (rtb + i) < rowTiles;

    v8f acc[ROWT][2];
#pragma unroll
    for (int i = 0; i < ROWT; ++i) {
        v8f a0 = {}, a1 = {};
        if (valid[i] && active && Csrc) {
            const int row0 = (rtb + i) << 4;
#pragma unroll
            for (int r = 0; r < 8; ++r)
                a0[r] = Csrc[(size_t)(row0 + r + hi * 8) * J + col0];
            if constexpr (NCT == 2) {
#pragma unroll
                for (int r = 0; r < 8; ++r)
                    a1[r] = Csrc[(size_t)(row0 + r + hi * 8) * J + col1];
            }
        }
        acc[i][0] = a0;
        acc[i][1] = a1;
    }

#pragma unroll 2
    for (int kc = 0; kc < K; kc += 32) {
        __syncthreads();
        // Stage W chunk: thread loads rows k0,k0+1 x 4 cols (2x b128), packs
        // f16 pairs (lo=k0, hi=k0+1) into dwords -> 4 ds_store_b32.
        constexpr int JG = J / 4;                // col groups of 4
        constexpr int ITER = (J * 32) / (256 * 8);
#pragma unroll
        for (int it = 0; it < ITER; ++it) {
            const int idx = tid + it * 256;
            const int jg = idx % JG, kg = idx / JG;
            const int j0 = jg * 4, k0 = kg * 2;
            v4f wa = *(const v4f*)(W + (size_t)(kc + k0) * J + j0);
            v4f wb = *(const v4f*)(W + (size_t)(kc + k0 + 1) * J + j0);
#pragma unroll
            for (int q = 0; q < 4; ++q) {
                unsigned lo = __builtin_bit_cast(unsigned short, (_Float16)wa[q]);
                unsigned hh = __builtin_bit_cast(unsigned short, (_Float16)wb[q]);
                sWu[(j0 + q) * 16 + (k0 >> 1)] = lo | (hh << 16);
            }
        }
        __syncthreads();

        if (active) {                            // wave-uniform branch
            // B fragments: loaded once, reused by all ROWT A-fragments.
            v16h bf0, bf1;
#pragma unroll
            for (int v = 0; v < 8; ++v) {
                const int kb = ((v < 4) ? (2 * v) : (16 + 2 * (v - 4))) + hi * 8;
                bf0[2 * v]     = sW[col0 * 32 + kb];
                bf0[2 * v + 1] = sW[col0 * 32 + kb + 1];
                if constexpr (NCT == 2) {
                    bf1[2 * v]     = sW[col1 * 32 + kb];
                    bf1[2 * v + 1] = sW[col1 * 32 + kb + 1];
                }
            }

#pragma unroll
            for (int i = 0; i < ROWT; ++i) {
                if (valid[i]) {                  // block-uniform branch
                    const float* xr = X + (size_t)(((rtb + i) << 4) + ln) * K;
                    v4f r0a = *(const v4f*)(xr + kc + hi * 8);
                    v4f r0b = *(const v4f*)(xr + kc + hi * 8 + 4);
                    v4f r1a = *(const v4f*)(xr + kc + hi * 8 + 16);
                    v4f r1b = *(const v4f*)(xr + kc + hi * 8 + 20);
                    v16h af;
#pragma unroll
                    for (int q = 0; q < 4; ++q) {
                        af[q]      = (_Float16)r0a[q];
                        af[4 + q]  = (_Float16)r0b[q];
                        af[8 + q]  = (_Float16)r1a[q];
                        af[12 + q] = (_Float16)r1b[q];
                    }
                    acc[i][0] = __builtin_amdgcn_wmma_f32_16x16x32_f16(
                        false, af, false, bf0, (short)0, acc[i][0], false, false);
                    if constexpr (NCT == 2) {
                        acc[i][1] = __builtin_amdgcn_wmma_f32_16x16x32_f16(
                            false, af, false, bf1, (short)0, acc[i][1], false, false);
                    }
                }
            }
        }
    }

    if (active) {
        const float bv0 = bias ? bias[col0] : 0.0f;
        const float bv1 = (NCT == 2 && bias) ? bias[col1] : 0.0f;
#pragma unroll
        for (int i = 0; i < ROWT; ++i) {
            if (valid[i]) {
                const int row0 = (rtb + i) << 4;
#pragma unroll
                for (int r = 0; r < 8; ++r)
                    Y[(size_t)(row0 + r + hi * 8) * J + col0] = acc[i][0][r] + bv0;
                if constexpr (NCT == 2) {
#pragma unroll
                    for (int r = 0; r < 8; ++r)
                        Y[(size_t)(row0 + r + hi * 8) * J + col1] = acc[i][1][r] + bv1;
                }
            }
        }
    }
}

// ---------------------------------------------------------------------------
// proj[n] = dot(t[n,:], h2[n,:]); graph-accumulate sums + counts.
// One wave per node, wave32 shuffle reduce.
// ---------------------------------------------------------------------------
__global__ __launch_bounds__(256) void k_dot_reduce(
    const float* __restrict__ t, const float* __restrict__ h2,
    const int* __restrict__ batch, float* __restrict__ gsum,
    float* __restrict__ gcnt, int n)
{
    int wave = threadIdx.x >> 5, lane = threadIdx.x & 31;
    int node = blockIdx.x * 8 + wave;
    if (node >= n) return;
    const float* tp = t  + (size_t)node * HDIM;
    const float* hp = h2 + (size_t)node * HDIM;
    float s = 0.0f;
#pragma unroll
    for (int j = 0; j < HDIM / 32; ++j)
        s = fmaf(tp[lane + j * 32], hp[lane + j * 32], s);
#pragma unroll
    for (int off = 16; off > 0; off >>= 1)
        s += __shfl_down(s, off, 32);
    if (lane == 0) {
        int g = batch[node];
        unsafeAtomicAdd(&gsum[g], s);
        unsafeAtomicAdd(&gcnt[g], 1.0f);
    }
}

__global__ void k_final(const float* __restrict__ gsum,
                        const float* __restrict__ gcnt,
                        float* __restrict__ out)
{
    int g = threadIdx.x;
    if (g < GDIM) out[g] = gsum[g] / fmaxf(gcnt[g], 1.0f);
}

// ---------------------------------------------------------------------------
extern "C" void kernel_launch(void* const* d_in, const int* in_sizes, int n_in,
                              void* d_out, int out_size, void* d_ws, size_t ws_size,
                              hipStream_t stream) {
    const float* x     = (const float*)d_in[0];
    const float* pos   = (const float*)d_in[1];
    const int*   ei    = (const int*)  d_in[2];
    const float* ea    = (const float*)d_in[3];
    const int*   batch = (const int*)  d_in[4];
    const float* Wf1   = (const float*)d_in[5];
    const float* b1    = (const float*)d_in[6];
    const float* Wr1   = (const float*)d_in[7];
    const float* Wo1   = (const float*)d_in[8];
    const float* Ws1   = (const float*)d_in[9];
    const float* Wf2   = (const float*)d_in[10];
    const float* b2    = (const float*)d_in[11];
    const float* Wr2   = (const float*)d_in[12];
    const float* Wo2   = (const float*)d_in[13];
    const float* Ws2   = (const float*)d_in[14];
    const float* Wp    = (const float*)d_in[15];
    float* out = (float*)d_out;

    const int N = in_sizes[4];          // batch has N entries
    const int E = in_sizes[3] / RDIM;   // edge_attr is E x 8

    // Workspace layout (floats)
    float* sh    = (float*)d_ws;                  // E*4
    float* feats = sh    + (size_t)E * 4;         // N*64 (feats1, reused feats2)
    float* agg   = feats + (size_t)N * CDIM;      // N*256 (both layers; reused as t)
    float* h     = agg   + (size_t)N * HDIM;      // N*256
    float* h2    = h     + (size_t)N * HDIM;      // N*256
    float* gsum  = h2    + (size_t)N * HDIM;      // 16
    float* gcnt  = gsum  + GDIM;                  // 16

    const int rowTiles = N / 16;                  // 50000/16 = 3125 exact
    constexpr int ROWT = 4;
    const int gemmBlocks = (rowTiles + ROWT - 1) / ROWT;   // 782

    // Edge geometry
    k_edge_sh<<<(E + 255) / 256, 256, 0, stream>>>(pos, ei, sh, E);

    // ---- Layer 1 ----
    k_feats1<<<(N * CDIM + 255) / 256, 256, 0, stream>>>(x, Wf1, b1, feats, N * CDIM);
    hipMemsetAsync(agg, 0, (size_t)N * HDIM * sizeof(float), stream);
    k_scatter<<<(E + 7) / 8, 256, 0, stream>>>(feats, ea, Wr1, sh, ei, ei + E, agg, E);
    k_gemm_wmma<HDIM, 2, ROWT><<<gemmBlocks, 256, 0, stream>>>(agg, Wo1, nullptr, nullptr, h, rowTiles);
    k_skip4<<<(N * HDIM + 255) / 256, 256, 0, stream>>>(x, Ws1, h, N * HDIM);

    // ---- Layer 2 ----
    k_gemm_wmma<CDIM, 1, ROWT><<<gemmBlocks, 256, 0, stream>>>(h, Wf2, b2, nullptr, feats, rowTiles);
    hipMemsetAsync(agg, 0, (size_t)N * HDIM * sizeof(float), stream);
    k_scatter<<<(E + 7) / 8, 256, 0, stream>>>(feats, ea, Wr2, sh, ei, ei + E, agg, E);
    k_gemm_wmma<HDIM, 2, ROWT><<<gemmBlocks, 256, 0, stream>>>(agg, Wo2, nullptr, nullptr, h2, rowTiles);
    k_gemm_wmma<HDIM, 2, ROWT><<<gemmBlocks, 256, 0, stream>>>(h,   Ws2, nullptr, h2,     h2, rowTiles); // += skip

    // ---- Bilinear readout: t = h2 @ W_proj; proj = rowdot(t, h2); graph mean ----
    k_gemm_wmma<HDIM, 2, ROWT><<<gemmBlocks, 256, 0, stream>>>(h2, Wp, nullptr, nullptr, agg /*t*/, rowTiles);
    hipMemsetAsync(gsum, 0, 2 * GDIM * sizeof(float), stream);
    k_dot_reduce<<<(N + 7) / 8, 256, 0, stream>>>(agg, h2, batch, gsum, gcnt, N);
    k_final<<<1, 32, 0, stream>>>(gsum, gcnt, out);
}
